// unrolled_block_sc_40759239639867
// MI455X (gfx1250) — matrix operand order repaired and checked by reading, not compiled, not run on hardware
//
#include <hip/hip_runtime.h>
#include <math.h>

typedef __bf16 bf16;
typedef __attribute__((ext_vector_type(16))) __bf16 v16bf;
typedef __attribute__((ext_vector_type(8)))  float  v8f;
typedef __attribute__((ext_vector_type(4)))  int    v4i;

#define NXY 65536   // 256*256
#define NB  4

#if __has_builtin(__builtin_amdgcn_global_load_async_to_lds_b128) && \
    __has_builtin(__builtin_amdgcn_s_wait_asynccnt)
#define USE_ASYNC_LDS 1
#else
#define USE_ASYNC_LDS 0
#endif

#define AS_GLOBAL __attribute__((address_space(1)))
#define AS_LDS    __attribute__((address_space(3)))

// ============================================================================
// Centered ortho 256-point FFT along a strided line.
// F_c(x)[k] = (-1)^k * FFT((-1)^n x[n])[k]   (exact for N=256 since N/2 even)
// scale 1/16 per 1-D pass (ortho).
// ============================================================================
__global__ void fft256_centered(float2* __restrict__ data, int lines_per_img,
                                long img_stride, long line_stride, long elem_stride,
                                int inverse)
{
  __shared__ float2 s[256];
  int l = blockIdx.x;
  float2* p = data + (long)(l / lines_per_img) * img_stride
                   + (long)(l % lines_per_img) * line_stride;
  int t = threadIdx.x; // 128 threads, 4 waves
#pragma unroll
  for (int e = 0; e < 2; ++e) {
    int n = t + (e << 7);
    float2 v = p[(long)n * elem_stride];
    float sg = (n & 1) ? -1.f : 1.f;
    s[__brev((unsigned)n) >> 24] = make_float2(v.x * sg, v.y * sg);
  }
  __syncthreads();
  float dir = inverse ? 1.f : -1.f;
  for (int stage = 0; stage < 8; ++stage) {
    int half = 1 << stage;
    int pos  = t & (half - 1);
    int i0   = ((t >> stage) << (stage + 1)) + pos;
    int i1   = i0 + half;
    float ang = dir * 3.14159265358979f * (float)pos / (float)half;
    float sw, cw; __sincosf(ang, &sw, &cw);
    float2 a = s[i0], b = s[i1];
    float2 tb = make_float2(b.x * cw - b.y * sw, b.x * sw + b.y * cw);
    s[i0] = make_float2(a.x + tb.x, a.y + tb.y);
    s[i1] = make_float2(a.x - tb.x, a.y - tb.y);
    __syncthreads();
  }
  const float scale = 0.0625f; // 1/16
#pragma unroll
  for (int e = 0; e < 2; ++e) {
    int k = t + (e << 7);
    float sg = (k & 1) ? -scale : scale;
    float2 v = s[k];
    p[(long)k * elem_stride] = make_float2(v.x * sg, v.y * sg);
  }
}

// Centered ortho DFT-4 along the batch axis (for the prox's full 4-D FFT).
__global__ void batchdft4_centered(float2* __restrict__ d, int inverse)
{
  int idx = blockIdx.x * blockDim.x + threadIdx.x;
  if (idx >= NXY) return;
  float2 x0 = d[idx];
  float2 x1 = d[idx + NXY];
  float2 x2 = d[idx + 2 * NXY];
  float2 x3 = d[idx + 3 * NXY];
  // input sign (-1)^b
  x1.x = -x1.x; x1.y = -x1.y; x3.x = -x3.x; x3.y = -x3.y;
  float si = inverse ? 1.f : -1.f; // W = si*i
  float2 ix1 = make_float2(-si * x1.y, si * x1.x);
  float2 ix3 = make_float2(-si * x3.y, si * x3.x);
  float2 y0 = make_float2(x0.x + x1.x + x2.x + x3.x, x0.y + x1.y + x2.y + x3.y);
  float2 y1 = make_float2(x0.x + ix1.x - x2.x - ix3.x, x0.y + ix1.y - x2.y - ix3.y);
  float2 y2 = make_float2(x0.x - x1.x + x2.x - x3.x, x0.y - x1.y + x2.y - x3.y);
  float2 y3 = make_float2(x0.x - ix1.x - x2.x + ix3.x, x0.y - ix1.y - x2.y + ix3.y);
  const float sc = 0.5f; // ortho 1/sqrt(4)
  // output sign (-1)^k
  d[idx]           = make_float2(y0.x * sc,  y0.y * sc);
  d[idx + NXY]     = make_float2(-y1.x * sc, -y1.y * sc);
  d[idx + 2 * NXY] = make_float2(y2.x * sc,  y2.y * sc);
  d[idx + 3 * NXY] = make_float2(-y3.x * sc, -y3.y * sc);
}

// ============================================================================
// Reductions for the (analytically collapsed) Armijo line search.
// S = sum mask*|Fx-b|^2, T = sum (1-mask)*|b|^2
// ============================================================================
__global__ void reduce_ST_kernel(const float2* __restrict__ Fx, const float2* __restrict__ bm,
                                 const unsigned char* __restrict__ mask, float* __restrict__ scal)
{
  int idx = blockIdx.x * 256 + threadIdx.x; // over 4*65536
  float dS = 0.f, dT = 0.f;
  unsigned char m = mask[idx & (NXY - 1)];
  float2 f = Fx[idx], bv = bm[idx];
  if (m) { float dx = f.x - bv.x, dy = f.y - bv.y; dS = dx * dx + dy * dy; }
  else   { dT = bv.x * bv.x + bv.y * bv.y; }
  for (int o = 16; o; o >>= 1) { dS += __shfl_down(dS, o, 32); dT += __shfl_down(dT, o, 32); }
  __shared__ float ss[8], st2[8];
  int wv = threadIdx.x >> 5, ln = threadIdx.x & 31;
  if (!ln) { ss[wv] = dS; st2[wv] = dT; }
  __syncthreads();
  if (threadIdx.x == 0) {
    float a = 0.f, b = 0.f;
    for (int i = 0; i < 8; ++i) { a += ss[i]; b += st2[i]; }
    scal[16 + 2 * blockIdx.x] = a;
    scal[17 + 2 * blockIdx.x] = b;
  }
}

// obj(a) = 0.5(S+T) - a*S + 0.5*a^2*S ; replicate lax.while_loop semantics exactly.
__global__ void finalize_alpha_kernel(float* __restrict__ scal)
{
  float S = 0.f, T = 0.f;
  for (int i = 0; i < 1024; ++i) { S += scal[16 + 2 * i]; T += scal[17 + 2 * i]; }
  float obj_x = 0.5f * (S + T);
  float gxn = S;
  float a_next = 1e-4f, a_used = 1e-4f;
  int it = 0; bool done = false;
  while (it < 250 && !done) {
    float a = a_next;
    float obj = 0.5f * (S + T) - a * S + 0.5f * a * a * S;
    done = obj < (obj_x - a * 0.9f * gxn);
    a_used = a;
    a_next = a * 0.9f;
    ++it;
  }
  scal[0] = S; scal[1] = T; scal[2] = a_used;
}

// R = mask ? (Fx - b) : 0   (in place on Fx)
__global__ void residual_kernel(float2* __restrict__ Fx, const float2* __restrict__ bm,
                                const unsigned char* __restrict__ mask)
{
  int idx = blockIdx.x * 256 + threadIdx.x;
  unsigned char m = mask[idx & (NXY - 1)];
  float2 f = Fx[idx], bv = bm[idx];
  Fx[idx] = m ? make_float2(f.x - bv.x, f.y - bv.y) : make_float2(0.f, 0.f);
}

// x1 = x - alpha*gx
__global__ void xupdate_kernel(const float2* __restrict__ x, const float2* __restrict__ gx,
                               float2* __restrict__ x1, const float* __restrict__ scal)
{
  int idx = blockIdx.x * 256 + threadIdx.x;
  float a = scal[2];
  float2 xv = x[idx], g = gx[idx];
  x1[idx] = make_float2(xv.x - a * g.x, xv.y - a * g.y);
}

// k-space replacement: z = mask ? b : z
__global__ void maskfill_kernel(float2* __restrict__ z, const float2* __restrict__ bm,
                                const unsigned char* __restrict__ mask)
{
  int idx = blockIdx.x * 256 + threadIdx.x;
  if (mask[idx & (NXY - 1)]) z[idx] = bm[idx];
}

// cat = [real ; imag] along H -> bf16 [4,512,256]
__global__ void cat_kernel(const float2* __restrict__ z, bf16* __restrict__ g)
{
  int idx = blockIdx.x * 256 + threadIdx.x; // 4*512*256
  int wq = idx & 255;
  int h  = (idx >> 8) & 511;
  int b  = idx >> 17;
  float v = (h < 256) ? z[(b * 256 + h) * 256 + wq].x
                      : z[(b * 256 + (h - 256)) * 256 + wq].y;
  g[idx] = (bf16)v;
}

// ============================================================================
// Weight repack: f32 OIHW -> bf16 [co][(kh*3+kw)*Cin + ci]  (tap-major K)
// ============================================================================
__global__ void pack_w_kernel(const float* __restrict__ w, bf16* __restrict__ wp,
                              int Cout, int Cin)
{
  int idx = blockIdx.x * 256 + threadIdx.x;
  int n = Cout * Cin * 9;
  if (idx >= n) return;
  int kw = idx % 3;
  int kh = (idx / 3) % 3;
  int ci = (idx / 9) % Cin;
  int co = idx / (9 * Cin);
  wp[(long)co * (9 * Cin) + (kh * 3 + kw) * Cin + ci] = (bf16)w[idx];
}

// ============================================================================
// WMMA 3x3 conv, NHWC bf16, f32 accumulate, bias+ReLU epilogue.
// Block: 128 threads = 4 waves; tile = 32 pixels x 32 couts (2x2 WMMA tiles).
// K = 9*Cin, stepped in 32s (tap-major). Optional dual input (skip concat)
// and nearest-2x upsample folded into the input addressing.
// Input tile is staged to LDS via GLOBAL_LOAD_ASYNC_TO_LDS_B128 when the
// toolchain exposes it (ASYNCcnt-tracked, no VGPR round-trip); boundary
// chunks are zero-filled with plain ds_store_b128 (disjoint, no hazard).
// ============================================================================
__global__ void __launch_bounds__(128)
conv3x3_wmma_kernel(const bf16* __restrict__ in0, int C0,
                    const bf16* __restrict__ in1, int C1,
                    const bf16* __restrict__ wp, const float* __restrict__ bias,
                    bf16* __restrict__ out,
                    int H, int W, int Cout, int ups)
{
  __shared__ bf16 smem[3 * 34 * 64]; // max Cin=64 -> 13056 B
  const int Cin = C0 + C1;
  const int K   = 9 * Cin;
  const int w0  = blockIdx.x * 32;
  const int h   = blockIdx.y;
  const int zz  = blockIdx.z;
  const int ng  = Cout >> 5;     // cout groups of 32
  const int cg  = zz % ng;
  const int b   = zz / ng;
  const int sH  = ups ? (H >> 1) : H;
  const int sW  = ups ? (W >> 1) : W;

  // ---- stage input tile: rows h-1..h+1, cols w0-1..w0+32, all Cin (NHWC) ----
  const int chunks = (3 * 34 * Cin) >> 3; // 128-bit chunks
  for (int ch = threadIdx.x; ch < chunks; ch += 128) {
    int pos = ch << 3;
    int ci  = pos % Cin;
    int rc  = pos / Cin;
    int c   = rc % 34;
    int r   = rc / 34;
    int ih  = h + r - 1;
    int iw  = w0 + c - 1;
    bf16* ldst = smem + (long)rc * Cin + ci;
    bool inb = (ih >= 0 && ih < H && iw >= 0 && iw < W);
    if (inb) {
      int sh = ups ? (ih >> 1) : ih;
      int sw = ups ? (iw >> 1) : iw;
      const bf16* src = (ci < C0)
          ? in0 + ((long)(b * sH + sh) * sW + sw) * C0 + ci
          : in1 + ((long)(b * sH + sh) * sW + sw) * C1 + (ci - C0);
#if USE_ASYNC_LDS
      __builtin_amdgcn_global_load_async_to_lds_b128(
          (AS_GLOBAL v4i*)(void*)src,
          (AS_LDS v4i*)(void*)ldst, 0, 0);
#else
      *(uint4*)ldst = *(const uint4*)src;
#endif
    } else {
      *(uint4*)ldst = make_uint4(0u, 0u, 0u, 0u);
    }
  }
#if USE_ASYNC_LDS
  __builtin_amdgcn_s_wait_asynccnt(0);
#endif
  __syncthreads();

  // ---- per-wave 16x16 WMMA tile ----
  const int wave = threadIdx.x >> 5;
  const int lane = threadIdx.x & 31;
  const int mi   = wave & 1;   // cout subtile
  const int ni   = wave >> 1;  // pixel subtile
  const int m    = lane & 15;
  const int hiA  = (lane < 16) ? 0 : 8;
  const int hiB  = (lane < 16) ? 0 : 8;
  const int co   = cg * 32 + mi * 16 + m;
  const bf16* wrow = wp + (long)co * K;
  __builtin_prefetch(wrow, 0, 3); // global_prefetch_b8: pull weight row toward L2

  v8f acc = {};
  for (int tap = 0; tap < 9; ++tap) {
    const int kh = tap / 3;
    const int kw = tap - kh * 3;
    const int cl = ni * 16 + (lane & 15) + kw; // LDS column for this lane/tap
    for (int cc = 0; cc < Cin; cc += 32) {
      const int kc = tap * Cin + cc;
      union { uint4 u[2]; v16bf v; } A;
      A.u[0] = *(const uint4*)(wrow + kc + hiA);        // K = kc+hiA+{0..7}
      A.u[1] = *(const uint4*)(wrow + kc + hiA + 16);   // K = kc+hiA+16+{0..7}
      union { uint4 u[2]; v16bf v; } Bf;
      const bf16* bp = smem + ((long)(kh * 34 + cl) * Cin + cc + hiB);
      Bf.u[0] = *(const uint4*)(bp);                    // K = cc+hiB+{0..7}
      Bf.u[1] = *(const uint4*)(bp + 16);               // K = cc+hiB+16+{0..7}
      acc = __builtin_amdgcn_wmma_f32_16x16x32_bf16(
          false, A.v, false, Bf.v, (short)0, acc, false, false);
    }
  }

  // ---- epilogue: bias + ReLU, packed bf16 NHWC store ----
  const int wpx    = w0 + ni * 16 + (lane & 15);
  const int cobase = cg * 32 + mi * 16 + ((lane < 16) ? 0 : 8);
  union { uint4 u; bf16 hh[8]; } st;
#pragma unroll
  for (int i = 0; i < 8; ++i) {
    float v = acc[i] + bias[cobase + i];
    st.hh[i] = (bf16)(v > 0.f ? v : 0.f);
  }
  *(uint4*)(out + ((long)(b * H + h) * W + wpx) * Cout + cobase) = st.u;
}

// first conv: 1 -> 32 channels (K too small for WMMA), bf16 in/out, ReLU
__global__ void conv1a_kernel(const bf16* __restrict__ g, const float* __restrict__ w,
                              const float* __restrict__ bias, bf16* __restrict__ out)
{
  int idx = blockIdx.x * 256 + threadIdx.x; // 4*512*256
  if (idx >= 4 * 512 * 256) return;
  int wq = idx & 255, h = (idx >> 8) & 511, b = idx >> 17;
  float pix[9];
#pragma unroll
  for (int kh = 0; kh < 3; ++kh)
#pragma unroll
    for (int kw = 0; kw < 3; ++kw) {
      int ih = h + kh - 1, iw = wq + kw - 1;
      pix[kh * 3 + kw] = (ih >= 0 && ih < 512 && iw >= 0 && iw < 256)
                             ? (float)g[(b * 512 + ih) * 256 + iw] : 0.f;
    }
  union { uint4 u[4]; bf16 hh[32]; } st;
  for (int co = 0; co < 32; ++co) {
    float acc = bias[co];
#pragma unroll
    for (int t = 0; t < 9; ++t) acc += w[co * 9 + t] * pix[t];
    st.hh[co] = (bf16)(acc > 0.f ? acc : 0.f);
  }
  uint4* dst = (uint4*)(out + (long)idx * 32);
#pragma unroll
  for (int i = 0; i < 4; ++i) dst[i] = st.u[i];
}

// 2x2 maxpool, NHWC bf16: [4,512,256,32] -> [4,256,128,32]
__global__ void maxpool_kernel(const bf16* __restrict__ in, bf16* __restrict__ out)
{
  int idx = blockIdx.x * 256 + threadIdx.x; // 4*256*128*32
  if (idx >= 4 * 256 * 128 * 32) return;
  int c  = idx & 31;
  int wq = (idx >> 5) & 127;
  int h  = (idx >> 12) & 255;
  int b  = idx >> 20;
  const bf16* p = in + ((long)(b * 512 + 2 * h) * 256 + 2 * wq) * 32 + c;
  float v0 = (float)p[0], v1 = (float)p[32];
  float v2 = (float)p[256 * 32], v3 = (float)p[256 * 32 + 32];
  float mx = v0 > v1 ? v0 : v1;
  mx = mx > v2 ? mx : v2;
  mx = mx > v3 ? mx : v3;
  out[idx] = (bf16)mx;
}

// final 1x1 conv 32->1, write complex-reinterleaved f32 output [B,1,256,256,2]
__global__ void conv4_kernel(const bf16* __restrict__ d, const float* __restrict__ w4,
                             const float* __restrict__ b4, float* __restrict__ out)
{
  int idx = blockIdx.x * 256 + threadIdx.x; // 4*512*256
  if (idx >= 4 * 512 * 256) return;
  int wq = idx & 255, h = (idx >> 8) & 511, b = idx >> 17;
  const bf16* p = d + (long)idx * 32;
  float acc = b4[0];
#pragma unroll
  for (int c = 0; c < 32; ++c) acc += w4[c] * (float)p[c];
  if (h < 256) out[((long)(b * 256 + h) * 256 + wq) * 2 + 0] = acc;
  else         out[((long)(b * 256 + (h - 256)) * 256 + wq) * 2 + 1] = acc;
}

// ============================================================================
extern "C" void kernel_launch(void* const* d_in, const int* in_sizes, int n_in,
                              void* d_out, int out_size, void* d_ws, size_t ws_size,
                              hipStream_t stream)
{
  const float2* x  = (const float2*)d_in[0];
  const float2* bm = (const float2*)d_in[1];
  const unsigned char* mask = (const unsigned char*)d_in[2];
  const float *w1a = (const float*)d_in[3],  *b1a = (const float*)d_in[4];
  const float *w1b = (const float*)d_in[5],  *b1b = (const float*)d_in[6];
  const float *w2a = (const float*)d_in[7],  *b2a = (const float*)d_in[8];
  const float *w2b = (const float*)d_in[9],  *b2b = (const float*)d_in[10];
  const float *wu  = (const float*)d_in[11], *bu  = (const float*)d_in[12];
  const float *w3a = (const float*)d_in[13], *b3a = (const float*)d_in[14];
  const float *w3b = (const float*)d_in[15], *b3b = (const float*)d_in[16];
  const float *w4  = (const float*)d_in[17], *b4  = (const float*)d_in[18];
  float* out = (float*)d_out;

  char* ws = (char*)d_ws;
  const size_t MB = 1024ull * 1024ull;
  float2* cplx0 = (float2*)(ws + 0);            // 2 MB  (Fx -> R -> gx)
  float2* cplx1 = (float2*)(ws + 2 * MB);       // 2 MB  (x1 -> kspace -> prox)
  float*  scal  = (float*)(ws + 4 * MB);        // 16 KB (partials + S,T,alpha)
  bf16*   wpB   = (bf16*)(ws + 4 * MB + 16 * 1024); // packed weights (<512 KB)
  bf16* wp1b = wpB;                  // 32x288
  bf16* wp2a = wp1b + 9216;          // 64x288
  bf16* wp2b = wp2a + 18432;         // 64x576
  bf16* wpu  = wp2b + 36864;         // 32x576
  bf16* wp3a = wpu  + 18432;         // 32x576
  bf16* wp3b = wp3a + 18432;         // 32x288
  bf16* G  = (bf16*)(ws + 5 * MB);    //  1 MB  cat [4,512,256]
  bf16* A  = (bf16*)(ws + 6 * MB);    // 32 MB  e1   [4,512,256,32]
  bf16* Bb = (bf16*)(ws + 38 * MB);   // 32 MB  e1a / up / d3b
  bf16* F  = (bf16*)(ws + 70 * MB);   //  8 MB  pool [4,256,128,32]
  bf16* D  = (bf16*)(ws + 78 * MB);   // 16 MB  e2a  [4,256,128,64]
  bf16* E  = (bf16*)(ws + 94 * MB);   // 16 MB  e2
  bf16* Hb = (bf16*)(ws + 110 * MB);  // 32 MB  d3a

  // ---------------- gradient step ----------------
  (void)hipMemcpyAsync(cplx0, x, (size_t)NB * NXY * sizeof(float2),
                       hipMemcpyDeviceToDevice, stream);
  // Fx = F_c(x): rows then cols per image
  fft256_centered<<<NB * 256, 128, 0, stream>>>(cplx0, 256, NXY, 256, 1, 0);
  fft256_centered<<<NB * 256, 128, 0, stream>>>(cplx0, 256, NXY, 1, 256, 0);
  // S, T reductions (deterministic two-stage)
  reduce_ST_kernel<<<1024, 256, 0, stream>>>(cplx0, bm, mask, scal);
  finalize_alpha_kernel<<<1, 1, 0, stream>>>(scal);
  // R = mask*(Fx - b); gx = F_c^H(R)
  residual_kernel<<<1024, 256, 0, stream>>>(cplx0, bm, mask);
  fft256_centered<<<NB * 256, 128, 0, stream>>>(cplx0, 256, NXY, 256, 1, 1);
  fft256_centered<<<NB * 256, 128, 0, stream>>>(cplx0, 256, NXY, 1, 256, 1);
  // x1 = x - alpha*gx
  xupdate_kernel<<<1024, 256, 0, stream>>>(x, cplx0, cplx1, scal);

  // ---------------- prox (full 4-D centered ortho FFT) ----------------
  fft256_centered<<<NB * 256, 128, 0, stream>>>(cplx1, 256, NXY, 256, 1, 0);
  fft256_centered<<<NB * 256, 128, 0, stream>>>(cplx1, 256, NXY, 1, 256, 0);
  batchdft4_centered<<<NXY / 256, 256, 0, stream>>>(cplx1, 0);
  maskfill_kernel<<<1024, 256, 0, stream>>>(cplx1, bm, mask);
  batchdft4_centered<<<NXY / 256, 256, 0, stream>>>(cplx1, 1);
  fft256_centered<<<NB * 256, 128, 0, stream>>>(cplx1, 256, NXY, 256, 1, 1);
  fft256_centered<<<NB * 256, 128, 0, stream>>>(cplx1, 256, NXY, 1, 256, 1);

  // ---------------- U-Net ----------------
  cat_kernel<<<2048, 256, 0, stream>>>(cplx1, G);

  pack_w_kernel<<<(32 * 32 * 9 + 255) / 256, 256, 0, stream>>>(w1b, wp1b, 32, 32);
  pack_w_kernel<<<(64 * 32 * 9 + 255) / 256, 256, 0, stream>>>(w2a, wp2a, 64, 32);
  pack_w_kernel<<<(64 * 64 * 9 + 255) / 256, 256, 0, stream>>>(w2b, wp2b, 64, 64);
  pack_w_kernel<<<(32 * 64 * 9 + 255) / 256, 256, 0, stream>>>(wu,  wpu,  32, 64);
  pack_w_kernel<<<(32 * 64 * 9 + 255) / 256, 256, 0, stream>>>(w3a, wp3a, 32, 64);
  pack_w_kernel<<<(32 * 32 * 9 + 255) / 256, 256, 0, stream>>>(w3b, wp3b, 32, 32);

  conv1a_kernel<<<2048, 256, 0, stream>>>(G, w1a, b1a, Bb);           // e1a
  conv3x3_wmma_kernel<<<dim3(8, 512, 4), 128, 0, stream>>>(           // e1
      Bb, 32, nullptr, 0, wp1b, b1b, A, 512, 256, 32, 0);
  maxpool_kernel<<<(4 * 256 * 128 * 32) / 256, 256, 0, stream>>>(A, F);
  conv3x3_wmma_kernel<<<dim3(4, 256, 8), 128, 0, stream>>>(           // e2a
      F, 32, nullptr, 0, wp2a, b2a, D, 256, 128, 64, 0);
  conv3x3_wmma_kernel<<<dim3(4, 256, 8), 128, 0, stream>>>(           // e2
      D, 64, nullptr, 0, wp2b, b2b, E, 256, 128, 64, 0);
  conv3x3_wmma_kernel<<<dim3(8, 512, 4), 128, 0, stream>>>(           // up (2x fused)
      E, 64, nullptr, 0, wpu, bu, Bb, 512, 256, 32, 1);
  conv3x3_wmma_kernel<<<dim3(8, 512, 4), 128, 0, stream>>>(           // d3a (concat fused)
      Bb, 32, A, 32, wp3a, b3a, Hb, 512, 256, 32, 0);
  conv3x3_wmma_kernel<<<dim3(8, 512, 4), 128, 0, stream>>>(           // d3b
      Hb, 32, nullptr, 0, wp3b, b3b, Bb, 512, 256, 32, 0);
  conv4_kernel<<<2048, 256, 0, stream>>>(Bb, w4, b4, out);
}
